// WeightOnlyInt8Linear_41644002902331
// MI455X (gfx1250) — compile-verified, matrix operand order checked by reading, not run on hardware
//
#include <hip/hip_runtime.h>

typedef __attribute__((ext_vector_type(16))) _Float16 v16h;
typedef __attribute__((ext_vector_type(8)))  _Float16 v8h;
typedef __attribute__((ext_vector_type(4)))  _Float16 v4h;
typedef __attribute__((ext_vector_type(8)))  float    v8f;
typedef __attribute__((ext_vector_type(4)))  int      v4i;

#define B_   4
#define S_   2048
#define IN_  4096
#define OUT_ 11008
#define M_TOTAL (B_ * S_)

#define MT 128      // M tile per block
#define NT 128      // N tile per block
#define KT 64       // K tile per iteration
#define LDS_STRIDE 72   // 64 + 8 halfs pad (144 B) -> conflict-free frag reads

#if __has_builtin(__builtin_amdgcn_global_load_async_to_lds_b128)
#define HAVE_ASYNC_LDS 1
typedef __attribute__((address_space(1))) v4i* gp_v4i;
typedef __attribute__((address_space(3))) v4i* lp_v4i;
#else
#define HAVE_ASYNC_LDS 0
#endif

union FragU { v16h v; v8h h[2]; };

__global__ __launch_bounds__(256)
void wo_int8_linear_wmma(const _Float16* __restrict__ x,
                         const int*      __restrict__ wq,
                         const _Float16* __restrict__ scale,
                         const _Float16* __restrict__ bias,
                         _Float16*       __restrict__ out)
{
    __shared__ _Float16 sA[MT][LDS_STRIDE];   // x tile [m][k]
    __shared__ _Float16 sW[NT][LDS_STRIDE];   // dequantized W tile [n][k]

    const int tid  = threadIdx.x;
    const int lane = tid & 31;                // wave32
    const int wid  = tid >> 5;                // 8 waves
    const int wm   = wid >> 2;                // 0..1 -> 64 M-rows per wave
    const int wn   = wid & 3;                 // 0..3 -> 32 N-cols per wave

    const int m0 = blockIdx.y * MT;
    const int n0 = blockIdx.x * NT;

    const int ln = lane & 15;                 // row/col within a 16x16 tile
    const int lh = (lane >> 4) * 8;           // K sub-offset per half-wave

    // Uniform bases (stay in SGPRs); per-lane offsets are 32-bit and
    // loop-invariant -> global_load saddr+voffset form, no 64-bit VALU adds.
    const _Float16* __restrict__ xb = x  + (size_t)m0 * IN_;
    const int*      __restrict__ wb = wq + (size_t)n0 * IN_;

    // Loop-invariant per-lane element offsets for the staging loops.
    int a_off[4];    // x tile: chunk = 8 halfs
    int a_row[4], a_kc[4];
    #pragma unroll
    for (int i = 0; i < 4; ++i) {
        int c    = tid + i * 256;             // 0..1023
        a_row[i] = c >> 3;                    // 8 chunks per row
        a_kc[i]  = (c & 7) << 3;
        a_off[i] = a_row[i] * IN_ + a_kc[i];
    }
    int w_off[8];    // W tile: chunk = 4 int32
    int w_col[8], w_kc[8];
    float w_scale[8];
    #pragma unroll
    for (int i = 0; i < 8; ++i) {
        int c    = tid + i * 256;             // 0..2047
        w_col[i] = c >> 4;                    // 16 chunks per output channel
        w_kc[i]  = (c & 15) << 2;
        w_off[i] = w_col[i] * IN_ + w_kc[i];
        w_scale[i] = (float)scale[n0 + w_col[i]];   // hoisted out of K loop
    }

    v8f acc[4][2] = {};

    for (int k0 = 0; k0 < IN_; k0 += KT) {
        __syncthreads();   // previous iteration's LDS readers must be done

        // ---- stage x tile: 128 rows x 64 halfs ----
#if HAVE_ASYNC_LDS
        #pragma unroll
        for (int i = 0; i < 4; ++i) {
            v4i* gsrc = (v4i*)(xb + a_off[i] + k0);        // drop const, retype
            v4i* ldst = (v4i*)&sA[a_row[i]][a_kc[i]];
            __builtin_amdgcn_global_load_async_to_lds_b128(
                (gp_v4i)gsrc, (lp_v4i)ldst, 0, 0);
        }
#else
        #pragma unroll
        for (int i = 0; i < 4; ++i)
            *(v8h*)&sA[a_row[i]][a_kc[i]] = *(const v8h*)(xb + a_off[i] + k0);
#endif

        // ---- stage + dequant W tile: 128 channels x 64 int8-in-int32 ----
        #pragma unroll
        for (int i = 0; i < 8; ++i) {
            v4i w = *(const v4i*)(wb + w_off[i] + k0);
            float s = w_scale[i];
            v4h hq;
            hq.x = (_Float16)((float)w.x * s);
            hq.y = (_Float16)((float)w.y * s);
            hq.z = (_Float16)((float)w.z * s);
            hq.w = (_Float16)((float)w.w * s);
            *(v4h*)&sW[w_col[i]][w_kc[i]] = hq;
        }

        // prefetch next K tile of weights while computing this one
        if (k0 + KT < IN_)
            __builtin_prefetch(wb + w_off[0] + k0 + KT, 0, 3);

#if HAVE_ASYNC_LDS
#if __has_builtin(__builtin_amdgcn_s_wait_asynccnt)
        __builtin_amdgcn_s_wait_asynccnt(0);
#else
        asm volatile("s_wait_asynccnt 0x0" ::: "memory");
#endif
#endif
        __syncthreads();

        // ---- two K=32 WMMA slices per staged tile ----
        #pragma unroll
        for (int ks = 0; ks < KT; ks += 32) {
            FragU a[4], b[2];
            #pragma unroll
            for (int mt = 0; mt < 4; ++mt) {
                const _Float16* p = &sA[wm * 64 + mt * 16 + ln][ks + lh];
                a[mt].h[0] = *(const v8h*)p;          // K = ks + lh .. +7
                a[mt].h[1] = *(const v8h*)(p + 16);   // K = ks + 16 + lh .. +7
            }
            #pragma unroll
            for (int nt = 0; nt < 2; ++nt) {
                const _Float16* p = &sW[wn * 32 + nt * 16 + ln][ks + lh];
                b[nt].h[0] = *(const v8h*)p;
                b[nt].h[1] = *(const v8h*)(p + 16);
            }
            #pragma unroll
            for (int mt = 0; mt < 4; ++mt)
                #pragma unroll
                for (int nt = 0; nt < 2; ++nt)
                    acc[mt][nt] = __builtin_amdgcn_wmma_f32_16x16x32_f16(
                        false, a[mt].v, false, b[nt].v,
                        (short)0, acc[mt][nt], false, false);
        }
    }

    // ---- epilogue: + bias, convert to fp16, store ----
    #pragma unroll
    for (int nt = 0; nt < 2; ++nt) {
        int col  = n0 + wn * 32 + nt * 16 + ln;
        float bc = (float)bias[col];
        #pragma unroll
        for (int mt = 0; mt < 4; ++mt) {
            #pragma unroll
            for (int v = 0; v < 8; ++v) {
                // C/D layout: VGPR v -> M = v (lanes 0-15) or v+8 (lanes 16-31)
                int row = m0 + wm * 64 + mt * 16 + (lh ? 8 : 0) + v;
                out[(size_t)row * OUT_ + col] = (_Float16)(acc[mt][nt][v] + bc);
            }
        }
    }
}

extern "C" void kernel_launch(void* const* d_in, const int* in_sizes, int n_in,
                              void* d_out, int out_size, void* d_ws, size_t ws_size,
                              hipStream_t stream) {
    const _Float16* x  = (const _Float16*)d_in[0];
    const int*      wq = (const int*)d_in[1];
    const _Float16* sc = (const _Float16*)d_in[2];
    const _Float16* bs = (const _Float16*)d_in[3];
    _Float16*       o  = (_Float16*)d_out;

    dim3 grid(OUT_ / NT, M_TOTAL / MT);   // (86, 64)
    dim3 block(256);
    hipLaunchKernelGGL(wo_int8_linear_wmma, grid, block, 0, stream,
                       x, wq, sc, bs, o);
}
